// Head2_39891656245686
// MI455X (gfx1250) — matrix operand order repaired and checked by reading, not compiled, not run on hardware
//
#include <hip/hip_runtime.h>
#include <hip/hip_bf16.h>
#include <math.h>

// ---------------------------------------------------------------------------
// NTM head addressing for MI455X (gfx1250, wave32, WMMA).
// Shapes: B=512, N=65536, C=1024, M=128.
// ---------------------------------------------------------------------------

typedef _Float16 half_t;
typedef __attribute__((ext_vector_type(16))) _Float16 v16h;
typedef __attribute__((ext_vector_type(8)))  float    v8f;

static constexpr int B = 512;
static constexpr int N = 65536;
static constexpr int C = 1024;
static constexpr int M = 128;
static constexpr int KSPLIT = 64;           // split-K chunks for read_vec GEMM
static constexpr float EPS_COS   = 1e-8f;
static constexpr float EPS_SHARP = 1e-16f;

__device__ __forceinline__ float softplusf(float x) {
    return fmaxf(x, 0.0f) + log1pf(expf(-fabsf(x)));
}

// wave32 shuffle reductions
__device__ __forceinline__ float wave_sum(float v) {
    #pragma unroll
    for (int off = 16; off; off >>= 1) v += __shfl_xor(v, off, 32);
    return v;
}
__device__ __forceinline__ float wave_max(float v) {
    #pragma unroll
    for (int off = 16; off; off >>= 1) v = fmaxf(v, __shfl_xor(v, off, 32));
    return v;
}

// ---------------------------------------------------------------------------
// Fused: mn[n] = ||memory[n,:]|| and row-major f16 copy memh[n][k].
// One wave per row; lane holds 4 consecutive elements.
// ---------------------------------------------------------------------------
__global__ void memcvt_kernel(const float* __restrict__ memory,
                              half_t* __restrict__ memh,
                              float* __restrict__ mn) {
    const int lane = threadIdx.x & 31;
    const int wave = threadIdx.x >> 5;
    const int row  = blockIdx.x * 8 + wave;        // block = 256 = 8 waves
    const float4 q = *(const float4*)(memory + (size_t)row * M + lane * 4);
    half_t* dst = memh + (size_t)row * M + lane * 4;
    dst[0] = (half_t)q.x; dst[1] = (half_t)q.y;
    dst[2] = (half_t)q.z; dst[3] = (half_t)q.w;
    float ss = q.x * q.x + q.y * q.y + q.z * q.z + q.w * q.w;
    ss = wave_sum(ss);
    if (lane == 0) mn[row] = sqrtf(ss);
}

// ---------------------------------------------------------------------------
// memT[m][n] = (f16)memory[n][m]  — B layout for the read_vec GEMM.
// 32x32 LDS tile transpose; grid (M/32, N/32), block (32, 8).
// ---------------------------------------------------------------------------
__global__ void memtrans_kernel(const float* __restrict__ memory,
                                half_t* __restrict__ memT) {
    __shared__ _Float16 tile[32][33];
    const int tx = threadIdx.x, ty = threadIdx.y;
    const int m0 = blockIdx.x * 32, n0 = blockIdx.y * 32;
    #pragma unroll
    for (int j = 0; j < 32; j += 8)
        tile[ty + j][tx] = (half_t)memory[(size_t)(n0 + ty + j) * M + m0 + tx];
    __syncthreads();
    #pragma unroll
    for (int j = 0; j < 32; j += 8)
        memT[(size_t)(m0 + ty + j) * N + n0 + tx] = tile[tx][ty + j];
}

// ---------------------------------------------------------------------------
// key = tanh(ctrl @ Wk + bk)   (B x C) @ (C x M) -> (B x M)
// WMMA f32 <- f16, 16x16 tile per wave, K loop over C.
// grid: (M/16, B/16), block: 32 (one wave)
// ---------------------------------------------------------------------------
__global__ void key_wmma_kernel(const float* __restrict__ ctrl,
                                const float* __restrict__ Wk,
                                const float* __restrict__ bk,
                                float* __restrict__ key_out) {
    const int lane = threadIdx.x & 31;
    const int r  = lane & 15;
    const int hi = lane >> 4;
    const int col0 = blockIdx.x * 16;
    const int row0 = blockIdx.y * 16;

    v8f c = {};
    for (int k0 = 0; k0 < C; k0 += 32) {
        v16h a, b;
        // A tile (16x32 f16): lane r holds row (row0+r); halves split K by 'hi'
        const float* arow = ctrl + (size_t)(row0 + r) * C + k0 + hi * 16;
        #pragma unroll
        for (int i = 0; i < 8; ++i) {
            a[i]     = (half_t)arow[i];
            a[8 + i] = (half_t)arow[8 + i];
        }
        // B tile (32x16 f16): lane holds column (col0+r), K range by 'hi'
        const float* bcol = Wk + (size_t)(k0 + hi * 16) * M + col0 + r;
        #pragma unroll
        for (int i = 0; i < 16; ++i) b[i] = (half_t)bcol[(size_t)i * M];
        c = __builtin_amdgcn_wmma_f32_16x16x32_f16(false, a, false, b,
                                                   (short)0, c, false, false);
    }
    const float bias = bk[col0 + r];
    #pragma unroll
    for (int v = 0; v < 8; ++v) {
        const int row = row0 + v + 8 * hi;       // C/D layout: M = v + 8*hi, N = r
        key_out[(size_t)row * M + col0 + r] = tanhf(c[v] + bias);
    }
}

// ---------------------------------------------------------------------------
// kn[b] = ||key[b,:]||  -> stats[b*8+6]        grid B, block 128
// ---------------------------------------------------------------------------
__global__ void keynorm_kernel(const float* __restrict__ key,
                               float* __restrict__ stats) {
    __shared__ float red[4];
    const int b = blockIdx.x, tid = threadIdx.x;
    float v = key[(size_t)b * M + tid];
    float ss = wave_sum(v * v);
    if ((tid & 31) == 0) red[tid >> 5] = ss;
    __syncthreads();
    if (tid == 0) stats[b * 8 + 6] = sqrtf(red[0] + red[1] + red[2] + red[3]);
}

// ---------------------------------------------------------------------------
// Per-row scalar gates: beta, gate, gamma, shift softmax.  grid B, block 256.
// stats layout per row: [beta, gate, gamma, s0, s1, s2, kn, pad]
// ---------------------------------------------------------------------------
__global__ void scalars_kernel(const float* __restrict__ ctrl,
                               const float* __restrict__ Wb,  const float* __restrict__ bb,
                               const float* __restrict__ Wg,  const float* __restrict__ bg,
                               const float* __restrict__ Ws,  const float* __restrict__ bs,
                               const float* __restrict__ Wgm, const float* __restrict__ bgm,
                               float* __restrict__ stats) {
    __shared__ float red[8][6];
    const int b = blockIdx.x, tid = threadIdx.x;
    const int lane = tid & 31, wid = tid >> 5;
    const float* x = ctrl + (size_t)b * C;
    float ab = 0.f, ag = 0.f, agm = 0.f, a0 = 0.f, a1 = 0.f, a2 = 0.f;
    for (int k = tid; k < C; k += 256) {
        const float xv = x[k];
        ab  += xv * Wb[k];
        ag  += xv * Wg[k];
        agm += xv * Wgm[k];
        a0  += xv * Ws[k * 3 + 0];
        a1  += xv * Ws[k * 3 + 1];
        a2  += xv * Ws[k * 3 + 2];
    }
    ab = wave_sum(ab); ag = wave_sum(ag); agm = wave_sum(agm);
    a0 = wave_sum(a0); a1 = wave_sum(a1); a2 = wave_sum(a2);
    if (lane == 0) {
        red[wid][0] = ab;  red[wid][1] = ag;  red[wid][2] = agm;
        red[wid][3] = a0;  red[wid][4] = a1;  red[wid][5] = a2;
    }
    __syncthreads();
    if (tid == 0) {
        float t[6] = {0, 0, 0, 0, 0, 0};
        #pragma unroll
        for (int w = 0; w < 8; ++w)
            #pragma unroll
            for (int j = 0; j < 6; ++j) t[j] += red[w][j];
        const float beta  = softplusf(t[0] + bb[0]);
        const float gate  = 1.0f / (1.0f + expf(-(t[1] + bg[0])));
        const float gamma = 1.0f + softplusf(t[2] + bgm[0]);
        float l0 = t[3] + bs[0], l1 = t[4] + bs[1], l2 = t[5] + bs[2];
        const float mx = fmaxf(l0, fmaxf(l1, l2));
        const float e0 = expf(l0 - mx), e1 = expf(l1 - mx), e2 = expf(l2 - mx);
        const float inv = 1.0f / (e0 + e1 + e2);
        float* st = stats + b * 8;
        st[0] = beta;  st[1] = gate;  st[2] = gamma;
        st[3] = e0 * inv; st[4] = e1 * inv; st[5] = e2 * inv;
    }
}

// ---------------------------------------------------------------------------
// logits[b,n] = beta[b] * (key[b,:] . memory[n,:]) / (kn[b]*mn[n] + eps)
// grid: (N/128, B/16), block 256 (8 waves, one 16x16 n-tile each).
// Key tile staged in LDS as f16; memory streamed as pre-converted f16 —
// the B operand of each WMMA is ONE aligned 32-byte v16h load.
// ---------------------------------------------------------------------------
__global__ void sim_wmma_kernel(const half_t* __restrict__ keyh_staging_unused,
                                const float* __restrict__ key,
                                const half_t* __restrict__ memh,
                                const float* __restrict__ stats,
                                const float* __restrict__ mn,
                                float* __restrict__ logits) {
    __shared__ _Float16 As[16][M];          // 4 KB: 16 key rows, full K=128
    const int tid  = threadIdx.x;
    const int lane = tid & 31, wave = tid >> 5;
    const int row0 = blockIdx.y * 16;
    for (int i = tid; i < 16 * M; i += 256)
        As[i >> 7][i & 127] = (half_t)key[(size_t)(row0 + (i >> 7)) * M + (i & 127)];
    __syncthreads();

    const int n0 = blockIdx.x * 128 + wave * 16;
    const int r  = lane & 15;
    const int hi = lane >> 4;
    const half_t* brow = memh + (size_t)(n0 + r) * M;    // B[k][n] = memh[n][k]

    v8f c = {};
    #pragma unroll
    for (int k0 = 0; k0 < M; k0 += 32) {
        v16h a;
        #pragma unroll
        for (int i = 0; i < 8; ++i) {
            a[i]     = As[r][k0 + hi * 16 + i];
            a[8 + i] = As[r][k0 + hi * 16 + 8 + i];
        }
        const v16h b = *(const v16h*)(brow + k0 + hi * 16);   // 32B aligned
        c = __builtin_amdgcn_wmma_f32_16x16x32_f16(false, a, false, b,
                                                   (short)0, c, false, false);
    }
    const float mncol = mn[n0 + r];
    #pragma unroll
    for (int v = 0; v < 8; ++v) {
        const int row = row0 + v + 8 * hi;
        const float beta = stats[row * 8 + 0];
        const float kn   = stats[row * 8 + 6];
        logits[(size_t)row * N + n0 + r] = beta * c[v] / (kn * mncol + EPS_COS);
    }
}

// ---------------------------------------------------------------------------
// Fused per-row pipeline: softmax -> gate blend -> circular conv -> sharpen.
// One workgroup per row; the entire N=65536 row lives in LDS (256 KB of the
// WGP's 320 KB).  grid B, block 1024 (32 waves), dynamic LDS (N+64)*4 bytes.
// gout holds logits on entry and the final head weights on exit.
// ---------------------------------------------------------------------------
__global__ void fused_row_kernel(float* __restrict__ gout,
                                 const float* __restrict__ prev,
                                 const float* __restrict__ stats) {
    extern __shared__ float smem[];
    float* s   = smem;          // N floats
    float* red = smem + N;      // 32+ floats for cross-wave reduction
    const int b = blockIdx.x, tid = threadIdx.x;
    const int lane = tid & 31, wid = tid >> 5;
    const float gate  = stats[b * 8 + 1];
    const float gamma = stats[b * 8 + 2];
    const float s0 = stats[b * 8 + 3], s1 = stats[b * 8 + 4], s2 = stats[b * 8 + 5];
    float* L = gout + (size_t)b * N;
    const float* P = prev + (size_t)b * N;

    // phase 1: load logits into LDS, row max
    float lmax = -3.402823466e38f;
    for (int i = tid; i < N; i += 1024) {
        const float v = L[i];
        s[i] = v;
        lmax = fmaxf(lmax, v);
    }
    lmax = wave_max(lmax);
    if (lane == 0) red[wid] = lmax;
    __syncthreads();
    if (wid == 0) { float v = wave_max(red[lane]); if (lane == 0) red[0] = v; }
    __syncthreads();
    const float rowmax = red[0];
    __syncthreads();

    // phase 2: exp + row sum (args <= 0 -> fast hardware exp is safe)
    float lsum = 0.f;
    for (int i = tid; i < N; i += 1024) {
        const float e = __expf(s[i] - rowmax);
        s[i] = e;
        lsum += e;
    }
    lsum = wave_sum(lsum);
    if (lane == 0) red[wid] = lsum;
    __syncthreads();
    if (wid == 0) { float v = wave_sum(red[lane]); if (lane == 0) red[0] = v; }
    __syncthreads();
    const float invsum = 1.0f / red[0];
    __syncthreads();

    // phase 3: gate blend with previous weights
    for (int i = tid; i < N; i += 1024)
        s[i] = gate * (s[i] * invsum) + (1.0f - gate) * P[i];
    __syncthreads();

    // phase 4: circular 3-tap conv + pow(gamma); write t to global, sum t
    float tsum = 0.f;
    for (int i = tid; i < N; i += 1024) {
        const int im = (i - 1) & (N - 1);
        const int ip = (i + 1) & (N - 1);
        const float wsv = s0 * s[im] + s1 * s[i] + s2 * s[ip];
        const float t = __powf(wsv, gamma);
        L[i] = t;
        tsum += t;
    }
    tsum = wave_sum(tsum);
    if (lane == 0) red[wid] = tsum;
    __syncthreads();
    if (wid == 0) { float v = wave_sum(red[lane]); if (lane == 0) red[0] = v; }
    __syncthreads();
    const float scale = 1.0f / (red[0] + EPS_SHARP);

    // phase 5: normalize in place (same thread touches same elements)
    for (int i = tid; i < N; i += 1024) L[i] *= scale;
}

// ---------------------------------------------------------------------------
// read_vec partials: (B x N) @ (N x M), split-K into KSPLIT chunks.
// grid: (KSPLIT, B/16, M/16), block 32.  Weights pre-scaled by 2^12 before
// f16 conversion (softmax-scale values ~1e-5 would denorm in f16).
// B operand comes from the transposed f16 copy: one v16h load per WMMA.
// ---------------------------------------------------------------------------
__global__ void readvec_wmma_kernel(const float* __restrict__ w,
                                    const half_t* __restrict__ memT,
                                    float* __restrict__ part) {
    const int lane = threadIdx.x & 31;
    const int r  = lane & 15;
    const int hi = lane >> 4;
    const int row0  = blockIdx.y * 16;
    const int col0  = blockIdx.z * 16;
    const int kbase = blockIdx.x * (N / KSPLIT);

    const half_t* bcol = memT + (size_t)(col0 + r) * N;  // B[k][col] = memT[col][k]
    v8f c = {};
    for (int k0 = kbase; k0 < kbase + N / KSPLIT; k0 += 32) {
        v16h a;
        const float* arow = w + (size_t)(row0 + r) * N + k0 + hi * 16;
        __builtin_prefetch(arow + 64, 0, 1);
        #pragma unroll
        for (int i = 0; i < 8; ++i) {
            a[i]     = (half_t)(arow[i]      * 4096.0f);
            a[8 + i] = (half_t)(arow[8 + i] * 4096.0f);
        }
        const v16h b = *(const v16h*)(bcol + k0 + hi * 16);  // 32B aligned
        c = __builtin_amdgcn_wmma_f32_16x16x32_f16(false, a, false, b,
                                                   (short)0, c, false, false);
    }
    float* dst = part + (size_t)blockIdx.x * (B * M);
    #pragma unroll
    for (int v = 0; v < 8; ++v) {
        const int row = row0 + v + 8 * hi;
        dst[(size_t)row * M + col0 + r] = c[v] * (1.0f / 4096.0f);
    }
}

// deterministic split-K reduction -> read_vec (B x M)
__global__ void reduce_partials_kernel(const float* __restrict__ part,
                                       float* __restrict__ out) {
    const int i = blockIdx.x * 256 + threadIdx.x;   // i < B*M
    float acc = 0.f;
    #pragma unroll 8
    for (int k = 0; k < KSPLIT; ++k) acc += part[(size_t)k * (B * M) + i];
    out[i] = acc;
}

// ---------------------------------------------------------------------------
extern "C" void kernel_launch(void* const* d_in, const int* in_sizes, int n_in,
                              void* d_out, int out_size, void* d_ws, size_t ws_size,
                              hipStream_t stream) {
    (void)in_sizes; (void)n_in; (void)out_size; (void)ws_size;

    const float* ctrl   = (const float*)d_in[0];
    const float* prevW[2] = {(const float*)d_in[1], (const float*)d_in[2]};
    const float* memory = (const float*)d_in[4];

    // per-head parameter sets (r = inputs 5..14, w = inputs 15..24)
    const float *Wk[2]  = {(const float*)d_in[5],  (const float*)d_in[15]};
    const float *bk[2]  = {(const float*)d_in[6],  (const float*)d_in[16]};
    const float *Wb[2]  = {(const float*)d_in[7],  (const float*)d_in[17]};
    const float *bb[2]  = {(const float*)d_in[8],  (const float*)d_in[18]};
    const float *Wg[2]  = {(const float*)d_in[9],  (const float*)d_in[19]};
    const float *bg[2]  = {(const float*)d_in[10], (const float*)d_in[20]};
    const float *Ws[2]  = {(const float*)d_in[11], (const float*)d_in[21]};
    const float *bs[2]  = {(const float*)d_in[12], (const float*)d_in[22]};
    const float *Wgm[2] = {(const float*)d_in[13], (const float*)d_in[23]};
    const float *bgm[2] = {(const float*)d_in[14], (const float*)d_in[24]};

    float* out = (float*)d_out;
    float* headOut[2] = {out, out + (size_t)B * N};
    float* readVec = out + 2 * (size_t)B * N;

    // workspace layout (floats):
    //   mn[N] | memh (N*M f16 = N*M/2 floats) | memT (M*N f16) | key_r[B*M]
    //   | key_w[B*M] | stats_r[B*8] | stats_w[B*8] | partials[KSPLIT*B*M]
    float* ws = (float*)d_ws;
    float* mn = ws;
    half_t* memh = (half_t*)(mn + N);
    half_t* memT = memh + (size_t)N * M;
    float* keyBase = (float*)(memT + (size_t)N * M);
    float* key[2]   = {keyBase, keyBase + B * M};
    float* stats[2] = {keyBase + 2 * B * M, keyBase + 2 * B * M + B * 8};
    float* partials = keyBase + 2 * B * M + 2 * B * 8;

    memcvt_kernel<<<N / 8, 256, 0, stream>>>(memory, memh, mn);
    memtrans_kernel<<<dim3(M / 32, N / 32), dim3(32, 8), 0, stream>>>(memory, memT);

    for (int h = 0; h < 2; ++h) {
        key_wmma_kernel<<<dim3(M / 16, B / 16), 32, 0, stream>>>(
            ctrl, Wk[h], bk[h], key[h]);
        keynorm_kernel<<<B, M, 0, stream>>>(key[h], stats[h]);
        scalars_kernel<<<B, 256, 0, stream>>>(
            ctrl, Wb[h], bb[h], Wg[h], bg[h], Ws[h], bs[h], Wgm[h], bgm[h],
            stats[h]);
        sim_wmma_kernel<<<dim3(N / 128, B / 16), 256, 0, stream>>>(
            nullptr, key[h], memh, stats[h], mn, headOut[h]);
        fused_row_kernel<<<B, 1024, (N + 64) * sizeof(float), stream>>>(
            headOut[h], prevW[h], stats[h]);
    }

    // read_vec = rweights @ memory (head 0 final weights already in d_out)
    readvec_wmma_kernel<<<dim3(KSPLIT, B / 16, M / 16), 32, 0, stream>>>(
        headOut[0], memT, partials);
    reduce_partials_kernel<<<(B * M) / 256, 256, 0, stream>>>(partials, readVec);
}
